// Top2Router_16879221473405
// MI455X (gfx1250) — matrix operand order repaired and checked by reading, not compiled, not run on hardware
//
#include <hip/hip_runtime.h>
#include <stdint.h>

// ---------------------------------------------------------------------------
// Top-2 router:  logits = x @ W^T + 0.01*noise ; gate = softmax ; top2(gate)
//   x     [4,4096,4096] f32 -> M = 16384 rows, K = 4096
//   W     [64,4096]     f32 -> N = 64
//   noise [4,4096,64]   f32
// Outputs (flat): top2_val [M,2] | top2_idx [M,2] (as f32) | gate [M,64]
// ws: partial logits (8 MiB) + pre-packed bf16 W fragments (1 MiB)
// ---------------------------------------------------------------------------

#define H_DIM   4096
#define E_DIM   64
#define M_TOTAL 16384
#define KSPLIT  2
#define KSLICE  (H_DIM / KSPLIT)   // 2048
#define KSTEPS  (KSLICE / 32)      // 64 wmma K-steps per slice
#define KSTEPS_TOTAL (H_DIM / 32)  // 128
#define WPB     4                  // waves per block (128 threads)

#define PART_FLOATS ((size_t)KSPLIT * M_TOTAL * E_DIM)  // 2M floats = 8 MiB

typedef __attribute__((ext_vector_type(16))) __bf16    v16bf;
typedef __attribute__((ext_vector_type(8)))  float     v8f;
typedef __attribute__((ext_vector_type(8)))  uint32_t  v8u;
typedef __attribute__((ext_vector_type(2)))  float     fvec2;

__device__ __forceinline__ uint32_t pack_bf16_hi(float f0, float f1) {
  // bf16(f0) in [15:0], bf16(f1) in [31:16] (truncation; residual goes to lo)
  return (__float_as_uint(f1) & 0xFFFF0000u) | (__float_as_uint(f0) >> 16);
}
__device__ __forceinline__ float bf16_residual(float f) {
  return f - __uint_as_float(__float_as_uint(f) & 0xFFFF0000u);
}

// ---------------------------------------------------------------------------
// Pack W into B-fragment order:  Wf[ksg][t][hilo][lane][v] : uint32
//   B frag (32x16 bf16): lane -> col n = lane&15, k = (lane>>4)*16 + v*2
// One thread per (ksg, t, lane, v): 128*4*32*8 = 131072 threads.
// ---------------------------------------------------------------------------
__global__ __launch_bounds__(256)
void pack_w(const float* __restrict__ W, uint32_t* __restrict__ Wf) {
  const int tid  = blockIdx.x * 256 + threadIdx.x;
  const int v    = tid & 7;
  const int lane = (tid >> 3) & 31;
  const int t    = (tid >> 8) & 3;
  const int ksg  = tid >> 10;                 // 0..127
  const int e    = t * 16 + (lane & 15);
  const int k    = ksg * 32 + ((lane >> 4) << 4) + (v << 1);

  const float f0 = W[(size_t)e * H_DIM + k];
  const float f1 = W[(size_t)e * H_DIM + k + 1];

  const size_t base = ((size_t)(ksg * 4 + t) * 2) * 256 + lane * 8 + v;
  Wf[base]       = pack_bf16_hi(f0, f1);                               // hi
  Wf[base + 256] = pack_bf16_hi(bf16_residual(f0), bf16_residual(f1)); // lo
}

// ---------------------------------------------------------------------------
// GEMM: each wave computes a 16x64 partial tile over a K-slice of 2048.
// c = Ahi*Bhi + Alo*Bhi + Ahi*Blo with 12 independent accumulators.
// ---------------------------------------------------------------------------
__global__ __launch_bounds__(WPB * 32)
void router_gemm(const float* __restrict__ x,
                 const uint32_t* __restrict__ Wf,
                 float* __restrict__ part) {
  const int lane = threadIdx.x & 31;
  const int wave = blockIdx.x * WPB + (threadIdx.x >> 5);
  const int ks   = wave & (KSPLIT - 1);      // K-slice id
  const int mt   = wave >> 1;                // M-tile id, 0..1023
  const int m0   = mt * 16;

  const int nrow = lane & 15;                // A row / B+C column within tile
  const int h    = lane >> 4;                // lane half

  const float* xrow = x + (size_t)(m0 + nrow) * H_DIM + ks * KSLICE;
  const uint32_t* wfrag =
      Wf + ((size_t)(ks * KSTEPS) * 4) * 2 * 256 + lane * 8;

  v8f cHH0 = {}, cHH1 = {}, cHH2 = {}, cHH3 = {};
  v8f cLH0 = {}, cLH1 = {}, cLH2 = {}, cLH3 = {};
  v8f cHL0 = {}, cHL1 = {}, cHL2 = {}, cHL3 = {};

  for (int kstep = 0; kstep < KSTEPS; ++kstep) {
    const int kb = kstep << 5;

    // ---- A fragment (16x32 bf16): k = h*16 + v*2 ----
    v8u ahu, alu;
#pragma unroll
    for (int v = 0; v < 8; ++v) {
      const int koff = (h << 4) + (v << 1);
      fvec2 f = __builtin_nontemporal_load((const fvec2*)(xrow + kb + koff));
      ahu[v] = pack_bf16_hi(f.x, f.y);
      alu[v] = pack_bf16_hi(bf16_residual(f.x), bf16_residual(f.y));
    }
    const v16bf a_hi = __builtin_bit_cast(v16bf, ahu);
    const v16bf a_lo = __builtin_bit_cast(v16bf, alu);

    // ---- pre-packed B fragments: Wf[ksg][t][hilo][lane][v] ----
    const uint32_t* wk = wfrag + (size_t)kstep * 4 * 2 * 256;

#define TILE_MMA(T, CHH, CLH, CHL)                                          \
    {                                                                       \
      const v16bf b_hi =                                                    \
          __builtin_bit_cast(v16bf, *(const v8u*)(wk + (T) * 512));         \
      const v16bf b_lo =                                                    \
          __builtin_bit_cast(v16bf, *(const v8u*)(wk + (T) * 512 + 256));   \
      CHH = __builtin_amdgcn_wmma_f32_16x16x32_bf16(false, a_hi, false,     \
                b_hi, (short)0, CHH, false, false);                         \
      CLH = __builtin_amdgcn_wmma_f32_16x16x32_bf16(false, a_lo, false,     \
                b_hi, (short)0, CLH, false, false);                         \
      CHL = __builtin_amdgcn_wmma_f32_16x16x32_bf16(false, a_hi, false,     \
                b_lo, (short)0, CHL, false, false);                         \
    }

    TILE_MMA(0, cHH0, cLH0, cHL0)
    TILE_MMA(1, cHH1, cLH1, cHL1)
    TILE_MMA(2, cHH2, cLH2, cHL2)
    TILE_MMA(3, cHH3, cLH3, cHL3)
#undef TILE_MMA
  }

  // ---- fold the three product accumulators and store partials ----
  const v8f c0 = cHH0 + cLH0 + cHL0;
  const v8f c1 = cHH1 + cLH1 + cHL1;
  const v8f c2 = cHH2 + cLH2 + cHL2;
  const v8f c3 = cHH3 + cLH3 + cHL3;

  float* out = part + (size_t)ks * M_TOTAL * E_DIM;
#pragma unroll
  for (int t = 0; t < 4; ++t) {
    const v8f c = (t == 0) ? c0 : (t == 1) ? c1 : (t == 2) ? c2 : c3;
#pragma unroll
    for (int v = 0; v < 8; ++v) {
      const int row = m0 + v + (h << 3);     // C layout: M = v + 8*half
      const int col = t * 16 + nrow;         // C layout: N = lane&15
      out[(size_t)row * E_DIM + col] = c[v];
    }
  }
}

// ---------------------------------------------------------------------------
// Epilogue: reduce K-slices, add noise, softmax over 64 experts, top-2.
// One row per thread.
// ---------------------------------------------------------------------------
__global__ __launch_bounds__(256)
void router_softmax_top2(const float* __restrict__ part,
                         const float* __restrict__ noise,
                         float* __restrict__ out) {
  const int row = blockIdx.x * blockDim.x + threadIdx.x;
  if (row >= M_TOTAL) return;

  const float4* p0 = (const float4*)(part + (size_t)row * E_DIM);
  const float4* p1 =
      (const float4*)(part + (size_t)M_TOTAL * E_DIM + (size_t)row * E_DIM);
  const float4* nz = (const float4*)(noise + (size_t)row * E_DIM);

  float l[E_DIM];
  float m = -1e30f;
#pragma unroll
  for (int i = 0; i < E_DIM / 4; ++i) {
    float4 a = p0[i], b = p1[i], n = nz[i];
    float4 v;
    v.x = a.x + b.x + 0.01f * n.x;
    v.y = a.y + b.y + 0.01f * n.y;
    v.z = a.z + b.z + 0.01f * n.z;
    v.w = a.w + b.w + 0.01f * n.w;
    l[4 * i + 0] = v.x; l[4 * i + 1] = v.y;
    l[4 * i + 2] = v.z; l[4 * i + 3] = v.w;
    m = fmaxf(m, fmaxf(fmaxf(v.x, v.y), fmaxf(v.z, v.w)));
  }

  float s = 0.0f;
#pragma unroll
  for (int e = 0; e < E_DIM; ++e) {
    l[e] = __expf(l[e] - m);
    s += l[e];
  }
  const float inv = 1.0f / s;

  float v1 = -1.0f, v2 = -1.0f;
  int   i1 = 0,     i2 = 0;
#pragma unroll
  for (int e = 0; e < E_DIM; ++e) {
    const float g = l[e] * inv;
    l[e] = g;
    if (g > v1)      { v2 = v1; i2 = i1; v1 = g; i1 = e; }
    else if (g > v2) { v2 = g;  i2 = e; }
  }

  // outputs: top2_val [M,2] | top2_idx [M,2] | gate [M,64]
  out[(size_t)row * 2 + 0] = v1;
  out[(size_t)row * 2 + 1] = v2;
  out[(size_t)(M_TOTAL * 2) + row * 2 + 0] = (float)i1;
  out[(size_t)(M_TOTAL * 2) + row * 2 + 1] = (float)i2;
  float4* g4 = (float4*)(out + (size_t)(M_TOTAL * 4) + (size_t)row * E_DIM);
#pragma unroll
  for (int i = 0; i < E_DIM / 4; ++i) {
    float4 v;
    v.x = l[4 * i + 0]; v.y = l[4 * i + 1];
    v.z = l[4 * i + 2]; v.w = l[4 * i + 3];
    g4[i] = v;
  }
}

// ---------------------------------------------------------------------------
extern "C" void kernel_launch(void* const* d_in, const int* in_sizes, int n_in,
                              void* d_out, int out_size, void* d_ws,
                              size_t ws_size, hipStream_t stream) {
  const float* x     = (const float*)d_in[0];
  const float* W     = (const float*)d_in[1];
  const float* noise = (const float*)d_in[2];
  float*    out  = (float*)d_out;
  float*    part = (float*)d_ws;                       // 8 MiB
  uint32_t* Wf   = (uint32_t*)((float*)d_ws + PART_FLOATS);  // 1 MiB

  pack_w<<<(KSTEPS_TOTAL * 4 * 32 * 8) / 256, 256, 0, stream>>>(W, Wf);

  const int total_waves = (M_TOTAL / 16) * KSPLIT;     // 2048
  router_gemm<<<total_waves / WPB, WPB * 32, 0, stream>>>(x, Wf, part);
  router_softmax_top2<<<M_TOTAL / 256, 256, 0, stream>>>(part, noise, out);
}
